// AttentionLayer_45389214384306
// MI455X (gfx1250) — compile-verified, matrix operand order checked by reading, not run on hardware
//
#include <hip/hip_runtime.h>

// ---------------- CDNA5 (gfx1250) types ----------------
typedef __attribute__((ext_vector_type(16))) __bf16 v16bf;
typedef __attribute__((ext_vector_type(8)))  __bf16 v8bf;
typedef __attribute__((ext_vector_type(8)))  float  v8f;

union FragBF { v16bf v; v8bf h[2]; };

#define kB  16
#define kT  2048
#define kD  256
#define kBT (kB * kT)
#define SROW 2064   // 2048 + 16 pad to stagger LDS banks across rows

// ---------------------------------------------------------------------------
// Kernel 1: h = h1 + h2 (f32->bf16), m = tanh(h) (bf16). 8 elems/thread.
// ---------------------------------------------------------------------------
__global__ __launch_bounds__(256)
void prep_kernel(const float* __restrict__ in, __bf16* __restrict__ h_bf,
                 __bf16* __restrict__ m_bf) {
  size_t idx = ((size_t)blockIdx.x * 256 + threadIdx.x) * 8;
  size_t bt = idx / kD;
  int d = (int)(idx % kD);
  const float* src = in + bt * (2 * kD) + d;
  v8bf hv, mv;
#pragma unroll
  for (int i = 0; i < 8; ++i) {
    float h = src[i] + src[kD + i];
    hv[i] = (__bf16)h;
    mv[i] = (__bf16)tanhf(h);
  }
  *(v8bf*)(h_bf + idx) = hv;
  *(v8bf*)(m_bf + idx) = mv;
}

// ---------------------------------------------------------------------------
// Kernel 2: W (256x256 f32) -> bf16
// ---------------------------------------------------------------------------
__global__ __launch_bounds__(256)
void wconv_kernel(const float* __restrict__ W, __bf16* __restrict__ Wbf) {
  int i = blockIdx.x * 256 + threadIdx.x;
  Wbf[i] = (__bf16)W[i];
}

// ---------------------------------------------------------------------------
// Kernel 3: transpose h_bf [b][t][d] -> hT [b][d][t] via 64x64 LDS tiles
// ---------------------------------------------------------------------------
__global__ __launch_bounds__(256)
void transpose_kernel(const __bf16* __restrict__ h_bf, __bf16* __restrict__ hT) {
  __shared__ __bf16 tile[64][65];
  int blk = blockIdx.x;
  int b   = blk / (32 * 4);
  int r2  = blk % (32 * 4);
  int t0  = (r2 / 4) * 64;
  int d0  = (r2 % 4) * 64;
#pragma unroll
  for (int k = 0; k < 16; ++k) {
    int i = threadIdx.x + k * 256;
    int r = i >> 6, c = i & 63;
    tile[r][c] = h_bf[((size_t)(b * kT + t0 + r)) * kD + d0 + c];
  }
  __syncthreads();
#pragma unroll
  for (int k = 0; k < 16; ++k) {
    int i = threadIdx.x + k * 256;
    int c = i >> 6, r = i & 63;  // consecutive tid -> consecutive t (coalesced)
    hT[((size_t)(b * kD + d0 + c)) * kT + t0 + r] = tile[r][c];
  }
}

// ---------------------------------------------------------------------------
// Kernel 4: aw = relu(h @ W^T + b)  -> bf16.  One 16x16 tile per wave,
// K=256 via 8x v_wmma_f32_16x16x32_bf16 (NT pattern: both operands row-major).
// ---------------------------------------------------------------------------
__global__ __launch_bounds__(256)
void gemm_aw_kernel(const __bf16* __restrict__ h_bf, const __bf16* __restrict__ Wbf,
                    const float* __restrict__ bias, __bf16* __restrict__ aw_bf) {
  int wave = threadIdx.x >> 5;
  int L    = threadIdx.x & 31;
  int half = L >> 4, ln = L & 15, base = half * 8;
  int tileId  = blockIdx.x * 8 + wave;
  int t0 = (tileId >> 4) * 16;   // row tile over B*T
  int e0 = (tileId & 15) * 16;   // col tile over D
  const __bf16* arow = h_bf + (size_t)(t0 + ln) * kD;
  const __bf16* brow = Wbf  + (size_t)(e0 + ln) * kD;
  v8f acc = {};
#pragma unroll
  for (int k = 0; k < kD; k += 32) {
    FragBF a, bq;
    a.h[0]  = *(const v8bf*)(arow + k + base);
    a.h[1]  = *(const v8bf*)(arow + k + 16 + base);
    bq.h[0] = *(const v8bf*)(brow + k + base);
    bq.h[1] = *(const v8bf*)(brow + k + 16 + base);
    acc = __builtin_amdgcn_wmma_f32_16x16x32_bf16(false, a.v, false, bq.v,
                                                  (short)0, acc, false, false);
  }
  float bv = bias[e0 + ln];
#pragma unroll
  for (int v = 0; v < 8; ++v) {
    int M = v + half * 8;
    float r = acc[v] + bv;
    r = r > 0.f ? r : 0.f;
    aw_bf[(size_t)(t0 + M) * kD + e0 + ln] = (__bf16)r;
  }
}

// ---------------------------------------------------------------------------
// Kernel 5: fused scores + row-softmax.  One workgroup = 32 t-rows x full 2048
// columns of one batch (two 16-row WMMA blocks share every B fragment -> 2x
// arithmetic intensity).  S block lives in ~266KB dynamic LDS (CDNA5 WGP has
// 320KB).  Then cooperative row max/sum, write P transposed (PT[s][t]) bf16.
// ---------------------------------------------------------------------------
__global__ __launch_bounds__(256)
void scores_kernel(const __bf16* __restrict__ m_bf, const __bf16* __restrict__ aw_bf,
                   __bf16* __restrict__ PT) {
  extern __shared__ float smem[];
  float* S    = smem;                // [32][SROW]
  float* redp = smem + 32 * SROW;    // [32][8]
  float* rowv = redp + 256;          // [32]
  int b  = blockIdx.x >> 6;          // / (kT/32)
  int t0 = (blockIdx.x & 63) * 32;
  int wave = threadIdx.x >> 5;
  int L    = threadIdx.x & 31;
  int half = L >> 4, ln = L & 15, base = half * 8;

  // cache A fragments for both 16-row blocks (t0.. and t0+16..), all K=256
  const __bf16* arow0 = m_bf + (size_t)(b * kT + t0 + ln) * kD;
  const __bf16* arow1 = m_bf + (size_t)(b * kT + t0 + 16 + ln) * kD;
  FragBF a0[8], a1[8];
#pragma unroll
  for (int k = 0; k < 8; ++k) {
    a0[k].h[0] = *(const v8bf*)(arow0 + 32 * k + base);
    a0[k].h[1] = *(const v8bf*)(arow0 + 32 * k + 16 + base);
    a1[k].h[0] = *(const v8bf*)(arow1 + 32 * k + base);
    a1[k].h[1] = *(const v8bf*)(arow1 + 32 * k + 16 + base);
  }
  for (int i = 0; i < 16; ++i) {
    int s0 = (wave * 16 + i) * 16;
    const __bf16* brow = aw_bf + (size_t)(b * kT + s0 + ln) * kD;
    v8f acc0 = {}, acc1 = {};
#pragma unroll
    for (int k = 0; k < 8; ++k) {
      FragBF bq;
      bq.h[0] = *(const v8bf*)(brow + 32 * k + base);
      bq.h[1] = *(const v8bf*)(brow + 32 * k + 16 + base);
      acc0 = __builtin_amdgcn_wmma_f32_16x16x32_bf16(false, a0[k].v, false, bq.v,
                                                     (short)0, acc0, false, false);
      acc1 = __builtin_amdgcn_wmma_f32_16x16x32_bf16(false, a1[k].v, false, bq.v,
                                                     (short)0, acc1, false, false);
    }
#pragma unroll
    for (int v = 0; v < 8; ++v) {
      S[(v + half * 8) * SROW + s0 + ln]        = acc0[v];
      S[(16 + v + half * 8) * SROW + s0 + ln]   = acc1[v];
    }
  }
  __syncthreads();

  // row-softmax over 32 rows: 8 threads per row
  int r  = threadIdx.x >> 3;
  int l8 = threadIdx.x & 7;
  float mx = -3.0e38f;
  for (int s = l8; s < kT; s += 8) mx = fmaxf(mx, S[r * SROW + s]);
  redp[r * 8 + l8] = mx;
  __syncthreads();
  if (threadIdx.x < 32) {
    float m2 = -3.0e38f;
#pragma unroll
    for (int j = 0; j < 8; ++j) m2 = fmaxf(m2, redp[threadIdx.x * 8 + j]);
    rowv[threadIdx.x] = m2;
  }
  __syncthreads();
  float rmax = rowv[r];
  float sum = 0.f;
  for (int s = l8; s < kT; s += 8) {
    float e = __expf(S[r * SROW + s] - rmax);
    S[r * SROW + s] = e;
    sum += e;
  }
  redp[r * 8 + l8] = sum;
  __syncthreads();
  if (threadIdx.x < 32) {
    float t = 0.f;
#pragma unroll
    for (int j = 0; j < 8; ++j) t += redp[threadIdx.x * 8 + j];
    rowv[threadIdx.x] = 1.0f / t;
  }
  __syncthreads();
  float inv[32];
#pragma unroll
  for (int j = 0; j < 32; ++j) inv[j] = rowv[j];

  // write PT[b][s][t0..t0+31] : 64 contiguous bytes per s
  __bf16* outp = PT + (size_t)b * kT * kT + t0;
  for (int s = (int)threadIdx.x; s < kT; s += 256) {
#pragma unroll
    for (int c = 0; c < 4; ++c) {
      v8bf p;
#pragma unroll
      for (int j = 0; j < 8; ++j)
        p[j] = (__bf16)(S[(c * 8 + j) * SROW + s] * inv[c * 8 + j]);
      *(v8bf*)(outp + (size_t)s * kT + c * 8) = p;
    }
  }
}

// ---------------------------------------------------------------------------
// Kernel 6: context[b][d][s] = sum_t hT[d][t] * PT[s][t]  (K=2048)
// 4x4 register-blocked: each wave owns a 64x64 output block (16 acc tiles);
// per k-step, 8 fragment loads feed 16 WMMAs -> ~32 FLOP/byte from L2.
// + residual h[b][s][d] recomputed from the raw input, f32 output.
// ---------------------------------------------------------------------------
__global__ __launch_bounds__(256)
void context_kernel(const __bf16* __restrict__ hT, const __bf16* __restrict__ PT,
                    const float* __restrict__ in, float* __restrict__ out) {
  int wave = threadIdx.x >> 5;
  int L    = threadIdx.x & 31;
  int half = L >> 4, ln = L & 15, base = half * 8;
  int tileId = blockIdx.x * 8 + wave;         // 0..2047
  int b    = tileId >> 7;                     // /128
  int rem  = tileId & 127;
  int dbas = (rem >> 5) * 64;                 // 4 d-blocks of 64
  int sbas = (rem & 31) * 64;                 // 32 s-blocks of 64

  const __bf16* arow[4];
  const __bf16* brow[4];
#pragma unroll
  for (int i = 0; i < 4; ++i)
    arow[i] = hT + (size_t)(b * kD + dbas + i * 16 + ln) * kT;
#pragma unroll
  for (int j = 0; j < 4; ++j)
    brow[j] = PT + (size_t)b * kT * kT + (size_t)(sbas + j * 16 + ln) * kT;

  v8f acc[4][4] = {};
  for (int k = 0; k < kT; k += 32) {
    FragBF a[4], bq[4];
#pragma unroll
    for (int i = 0; i < 4; ++i) {
      a[i].h[0] = *(const v8bf*)(arow[i] + k + base);
      a[i].h[1] = *(const v8bf*)(arow[i] + k + 16 + base);
    }
#pragma unroll
    for (int j = 0; j < 4; ++j) {
      bq[j].h[0] = *(const v8bf*)(brow[j] + k + base);
      bq[j].h[1] = *(const v8bf*)(brow[j] + k + 16 + base);
    }
#pragma unroll
    for (int i = 0; i < 4; ++i)
#pragma unroll
      for (int j = 0; j < 4; ++j)
        acc[i][j] = __builtin_amdgcn_wmma_f32_16x16x32_bf16(
            false, a[i].v, false, bq[j].v, (short)0, acc[i][j], false, false);
  }

#pragma unroll
  for (int i = 0; i < 4; ++i) {
#pragma unroll
    for (int j = 0; j < 4; ++j) {
      int d0 = dbas + i * 16;
      int s0 = sbas + j * 16;
      const float* inrow = in + (size_t)(b * kT + s0 + ln) * (2 * kD) + d0;
#pragma unroll
      for (int v = 0; v < 8; ++v) {
        int M = v + half * 8;
        float hres = inrow[M] + inrow[kD + M];
        out[(size_t)(b * kD + d0 + M) * kT + s0 + ln] = acc[i][j][v] + hres;
      }
    }
  }
}

// ---------------------------------------------------------------------------
extern "C" void kernel_launch(void* const* d_in, const int* in_sizes, int n_in,
                              void* d_out, int out_size, void* d_ws, size_t ws_size,
                              hipStream_t stream) {
  (void)in_sizes; (void)n_in; (void)out_size; (void)ws_size;
  const float* in   = (const float*)d_in[0];
  const float* W    = (const float*)d_in[1];
  const float* bias = (const float*)d_in[2];
  float* out = (float*)d_out;

  char* ws = (char*)d_ws;
  size_t off = 0;
  __bf16* h_bf  = (__bf16*)(ws + off); off += (size_t)kBT * kD * 2;   // 16 MiB
  __bf16* m_bf  = (__bf16*)(ws + off); off += (size_t)kBT * kD * 2;   // 16 MiB
  __bf16* aw_bf = (__bf16*)(ws + off); off += (size_t)kBT * kD * 2;   // 16 MiB
  __bf16* hT_bf = (__bf16*)(ws + off); off += (size_t)kBT * kD * 2;   // 16 MiB
  __bf16* Wbf   = (__bf16*)(ws + off); off += (size_t)kD * kD * 2;    // 128 KiB
  off = (off + 255) & ~(size_t)255;
  __bf16* PT    = (__bf16*)(ws + off); off += (size_t)kB * kT * kT * 2; // 128 MiB

  prep_kernel<<<kBT * kD / (256 * 8), 256, 0, stream>>>(in, h_bf, m_bf);
  wconv_kernel<<<kD * kD / 256, 256, 0, stream>>>(W, Wbf);
  transpose_kernel<<<kB * (kT / 64) * (kD / 64), 256, 0, stream>>>(h_bf, hT_bf);
  gemm_aw_kernel<<<(kBT / 16) * (kD / 16) / 8, 256, 0, stream>>>(h_bf, Wbf, bias, aw_bf);
  size_t smem = (size_t)(32 * SROW + 256 + 32) * sizeof(float);  // ~266 KB (CDNA5 WGP: 320 KB)
  scores_kernel<<<kB * (kT / 32), 256, smem, stream>>>(m_bf, aw_bf, PT);
  context_kernel<<<kB * (kD / 64) * (kT / 64) / 8, 256, 0, stream>>>(hT_bf, PT, in, out);
}
